// HoMELayer_75256416961131
// MI455X (gfx1250) — matrix operand order repaired and checked by reading, not compiled, not run on hardware
//
#include <hip/hip_runtime.h>
#include <math.h>
#include <stdint.h>

// ---------------------------------------------------------------------------
// HoMELayer for MI455X (gfx1250, wave32, WMMA + TDM).
// Heavy GEMMs: v_wmma_f32_16x16x32_f16, operands pre-converted to f16 so each
// 128x32 LDS tile is a single 2D tensor_load_to_lds DMA (TENSORcnt tracked).
// All GEMM dims are multiples of 128 (M,N) and 32 (K): no edge handling.
// ---------------------------------------------------------------------------

typedef __attribute__((ext_vector_type(16))) _Float16 v16h;
typedef __attribute__((ext_vector_type(8)))  _Float16 v8h;
typedef __attribute__((ext_vector_type(8)))  float    v8f;

union Frag16 { v16h v; v8h h[2]; };

#define BSZ   1024      // batch
#define TBM   128       // block tile M (batch rows)
#define TBN   128       // block tile N
#define TBK   32        // block tile K (= one wmma K step)
#define BN_EPS 1e-5f

__device__ __forceinline__ float sigmoidf_(float x) { return 1.0f / (1.0f + expf(-x)); }

#if defined(__HIP_DEVICE_COMPILE__) && __has_builtin(__builtin_amdgcn_tensor_load_to_lds)
#define HAVE_TDM 1
#else
#define HAVE_TDM 0
#endif

#if HAVE_TDM
typedef unsigned int v4u __attribute__((ext_vector_type(4)));
typedef int          v4i __attribute__((ext_vector_type(4)));
typedef int          v8i __attribute__((ext_vector_type(8)));

// 2D TDM: move a 32(e)x128(row) f16 tile (row stride = stride_elems) from
// global to LDS (packed rows of 32 halves).  D# per ISA 08_async_tensor §8.
__device__ __forceinline__ void tdm_load_tile_32x128(unsigned lds_off,
                                                     const _Float16* gptr,
                                                     long stride_elems)
{
    unsigned long long ga = (unsigned long long)(uintptr_t)gptr;
    v4u g0;
    g0.x = 1u;                                   // count=1 (valid descriptor)
    g0.y = lds_off;                              // lds_addr (bytes)
    g0.z = (unsigned)ga;                         // global_addr[31:0]
    g0.w = (unsigned)((ga >> 32) & 0x01FFFFFFu)  // global_addr[56:32]
           | (2u << 30);                         // type=2 ("image")
    unsigned long long st = (unsigned long long)stride_elems;
    v8i g1;
    g1[0] = (int)(1u << 16);                     // data_size=1 (2 bytes)
    g1[1] = (int)(32u << 16);                    // tensor_dim0[15:0]=32
    g1[2] = (int)(128u << 16);                   // dim0 hi=0 | tensor_dim1=128
    g1[3] = (int)(32u << 16);                    // dim1 hi=0 | tile_dim0=32
    g1[4] = (int)128u;                           // tile_dim1=128, tile_dim2=0
    g1[5] = (int)(unsigned)(st & 0xFFFFFFFFu);   // tensor_dim0_stride lo
    g1[6] = (int)(unsigned)((st >> 32) & 0xFFFFu); // stride hi | dim1_stride=0
    g1[7] = 0;
    v4i z4 = {0, 0, 0, 0};
#if defined(__clang_major__) && (__clang_major__ >= 23)
    v8i z8 = {0, 0, 0, 0, 0, 0, 0, 0};
    __builtin_amdgcn_tensor_load_to_lds(g0, g1, z4, z4, z8, 0);
#else
    __builtin_amdgcn_tensor_load_to_lds(g0, g1, z4, z4, 0);
#endif
}
#endif

// ---------------------------------------------------------------------------
// GEMM: C[b, e, n] = relu( A_e[b, :K] @ W_e[n, :K]^T + bias_e[n] )
//   A  : f16 [Brows, lda], per-expert offset strideAe (elements)
//   Wt : f16 [E][N][K] (pre-transposed: K contiguous)
//   out: f32 to Cf (out16=0) or f16 to Ch (out16=1); E = gridDim.z
// 256 threads = 8 waves as 2x4; wave tile 64x32 = 4x2 wmma tiles.
// ---------------------------------------------------------------------------
__global__ __launch_bounds__(256)
void gemm_bias_relu(const _Float16* __restrict__ A,
                    const _Float16* __restrict__ Wt,
                    const float* __restrict__ bias,
                    float* __restrict__ Cf,
                    _Float16* __restrict__ Ch,
                    int K, int N, int lda, long strideAe, int out16)
{
    const int E  = gridDim.z;
    const int e  = blockIdx.z;
    const _Float16* Ae = A  + (size_t)e * strideAe;
    const _Float16* We = Wt + (size_t)e * N * K;
    const float*    be = bias + (size_t)e * N;
    const int row0 = blockIdx.y * TBM;
    const int col0 = blockIdx.x * TBN;

    __shared__ __align__(16) _Float16 lsA[TBM * TBK];   // [row][k], packed
    __shared__ __align__(16) _Float16 lsB[TBN * TBK];   // [n][k], packed

    const int tid  = threadIdx.x;
    const int lane = tid & 31;
    const int wave = tid >> 5;
    const int wm   = wave & 1;   // 0..1  -> 64 rows each
    const int wn   = wave >> 1;  // 0..3  -> 32 cols each

    const int lhalf = lane >> 4;         // 0 | 1 (half-wave)
    const int lmod  = lane & 15;
    const int aklo  = lhalf * 8;         // A frag K chunk base
    const int bkb   = lhalf * 8;         // B frag K chunk base

    const v8f vzero = {0.f, 0.f, 0.f, 0.f, 0.f, 0.f, 0.f, 0.f};
    v8f acc[4][2];
#pragma unroll
    for (int mt = 0; mt < 4; ++mt)
#pragma unroll
        for (int nt = 0; nt < 2; ++nt) acc[mt][nt] = vzero;

#if HAVE_TDM
    const unsigned lsA_off = (unsigned)(uintptr_t)&lsA[0];
    const unsigned lsB_off = (unsigned)(uintptr_t)&lsB[0];
#endif

    for (int k0 = 0; k0 < K; k0 += TBK) {
#if HAVE_TDM
        if (wave == 0) {
            tdm_load_tile_32x128(lsA_off, Ae + (size_t)row0 * lda + k0, lda);
            tdm_load_tile_32x128(lsB_off, We + (size_t)col0 * K  + k0, K);
            __builtin_amdgcn_s_wait_tensorcnt(0);
        }
#else
        // Fallback: synchronous f16 tile copy (no conversion needed).
#pragma unroll
        for (int i = 0; i < 16; ++i) {
            int idx = tid + i * 256;            // 0..4095
            int r = idx >> 5, c = idx & 31;
            lsA[idx] = Ae[(size_t)(row0 + r) * lda + (k0 + c)];
            lsB[idx] = We[(size_t)(col0 + r) * K + (k0 + c)];
        }
#endif
        __syncthreads();

        // A fragments: lane holds row (base+lmod); K = {aklo..+7, 16+aklo..+7}
        Frag16 afrag[4];
#pragma unroll
        for (int mt = 0; mt < 4; ++mt) {
            const _Float16* pa = &lsA[(wm * 64 + mt * 16 + lmod) * TBK];
            afrag[mt].h[0] = *(const v8h*)(pa + aklo);
            afrag[mt].h[1] = *(const v8h*)(pa + 16 + aklo);
        }
        // B fragments: lane = column (base+lmod); K = {bkb..+7, 16+bkb..+7}
        Frag16 bfrag[2];
#pragma unroll
        for (int nt = 0; nt < 2; ++nt) {
            const _Float16* pb = &lsB[(wn * 32 + nt * 16 + lmod) * TBK + bkb];
            bfrag[nt].h[0] = *(const v8h*)(pb);
            bfrag[nt].h[1] = *(const v8h*)(pb + 16);
        }
#pragma unroll
        for (int mt = 0; mt < 4; ++mt)
#pragma unroll
            for (int nt = 0; nt < 2; ++nt)
                acc[mt][nt] = __builtin_amdgcn_wmma_f32_16x16x32_f16(
                    false, afrag[mt].v, false, bfrag[nt].v,
                    (short)0, acc[mt][nt], false, false);
        __syncthreads();
    }

    // Epilogue: D layout -> VGPR r: M = r + 8*lhalf ; N = lmod
    const int rhalf = lhalf * 8;
#pragma unroll
    for (int mt = 0; mt < 4; ++mt) {
#pragma unroll
        for (int nt = 0; nt < 2; ++nt) {
            int n  = col0 + wn * 32 + nt * 16 + lmod;
            float bv = be[n];
#pragma unroll
            for (int r = 0; r < 8; ++r) {
                int m = row0 + wm * 64 + mt * 16 + rhalf + r;
                float val = fmaxf(acc[mt][nt][r] + bv, 0.f);
                size_t idx = ((size_t)m * E + e) * N + n;
                if (out16) Ch[idx] = (_Float16)val;
                else       Cf[idx] = val;
            }
        }
    }
}

// ---------------------------------------------------------------------------
// Weight transpose + f16 convert: Wt[e][n][k] = (f16) W[e][k][n]
// ---------------------------------------------------------------------------
__global__ void cvt_w_transpose(const float* __restrict__ W,
                                _Float16* __restrict__ Wt,
                                int K, int N, long total /* E*K*N */)
{
    long i = (long)blockIdx.x * 256 + threadIdx.x;
    if (i >= total) return;
    long e   = i / ((long)K * N);
    long rem = i - e * (long)K * N;
    int  n   = (int)(rem / K);
    int  k   = (int)(rem - (long)n * K);
    Wt[i] = (_Float16)W[(e * K + k) * (long)N + n];
}

// ---------------------------------------------------------------------------
// Rank-2 LoRA reduce: tvec[b,r] = sum_d v[b,d] * Af[d,r], r in {0,1}.
// concat mode: v[b,d] = x[task[d/512], b, d%512]  (x: [6,1024,512] f32)
// plain  mode: V is f16 [B, ldv]. One wave per batch row.
// ---------------------------------------------------------------------------
__global__ void lora_rank2_reduce(const _Float16* __restrict__ V, int ldv,
                                  const float* __restrict__ X,
                                  int tA, int tB, int tC, int concat,
                                  const float* __restrict__ Af,
                                  int D, float* __restrict__ tvec)
{
    int b = blockIdx.x;
    int lane = threadIdx.x;
    float s0 = 0.f, s1 = 0.f;
    for (int d = lane; d < D; d += 32) {
        float v;
        if (concat) {
            int seg = d >> 9, off = d & 511;
            int tt = (seg == 0) ? tA : ((seg == 1) ? tB : tC);
            v = X[((size_t)tt * BSZ + b) * 512 + off];
        } else {
            v = (float)V[(size_t)b * ldv + d];
        }
        s0 = fmaf(v, Af[2 * d + 0], s0);
        s1 = fmaf(v, Af[2 * d + 1], s1);
    }
#pragma unroll
    for (int o = 16; o > 0; o >>= 1) {
        s0 += __shfl_xor(s0, o, 32);
        s1 += __shfl_xor(s1, o, 32);
    }
    if (lane == 0) { tvec[2 * b] = s0; tvec[2 * b + 1] = s1; }
}

// out[b,d] = 2*sigmoid( tvec[b,0]*Bf[0,d] + tvec[b,1]*Bf[1,d] )   (Bf: [2,D])
template <typename T>
__global__ void lora_rank2_apply(const float* __restrict__ tvec,
                                 const float* __restrict__ Bf,
                                 T* __restrict__ out, int D, long total)
{
    long i = (long)blockIdx.x * 256 + threadIdx.x;
    if (i >= total) return;
    long b = i / D; int d = (int)(i - b * D);
    float z = tvec[2 * b] * Bf[d] + tvec[2 * b + 1] * Bf[D + d];
    out[i] = (T)(2.f * sigmoidf_(z));
}

// ---------------------------------------------------------------------------
// BatchNorm (training stats over B=1024) + SiLU, applied in place (f32).
// ---------------------------------------------------------------------------
__global__ void bn_stats(const float* __restrict__ y,
                         float* __restrict__ s, float* __restrict__ q, int C)
{
    int c = blockIdx.x * blockDim.x + threadIdx.x;
    if (c >= C) return;
    float ss = 0.f, qq = 0.f;
    for (int b = 0; b < BSZ; ++b) {
        float v = y[(size_t)b * C + c];
        ss += v; qq = fmaf(v, v, qq);
    }
    s[c] = ss; q[c] = qq;
}

__global__ void bn_apply_silu(float* __restrict__ y,
                              const float* __restrict__ s,
                              const float* __restrict__ q,
                              const float* __restrict__ gamma,
                              const float* __restrict__ beta,
                              int C, long total)
{
    long i = (long)blockIdx.x * 256 + threadIdx.x;
    if (i >= total) return;
    int c = (int)(i % C);
    float mu  = s[c] * (1.f / BSZ);
    float var = q[c] * (1.f / BSZ) - mu * mu;
    float yn  = (y[i] - mu) * rsqrtf(var + BN_EPS) * gamma[c] + beta[c];
    y[i] = yn * sigmoidf_(yn);
}

// ---------------------------------------------------------------------------
// Gate: out[b, 0:n] = softmax( in[b,:128] @ Wm[128,n] + bv[n] ) ; wave per row
// ---------------------------------------------------------------------------
__global__ void gate_softmax(const float* __restrict__ in,
                             const float* __restrict__ Wm,
                             const float* __restrict__ bv,
                             int n, float* __restrict__ out)
{
    int b = blockIdx.x, lane = threadIdx.x;
    float logit = -3.0e38f;
    if (lane < n) {
        float acc = bv[lane];
        const float* row = in + (size_t)b * 128;
        for (int k = 0; k < 128; ++k)
            acc = fmaf(row[k], Wm[(size_t)k * n + lane], acc);
        logit = acc;
    }
    float mx = logit;
#pragma unroll
    for (int o = 16; o > 0; o >>= 1) mx = fmaxf(mx, __shfl_xor(mx, o, 32));
    float ex = (lane < n) ? expf(logit - mx) : 0.f;
    float sm = ex;
#pragma unroll
    for (int o = 16; o > 0; o >>= 1) sm += __shfl_xor(sm, o, 32);
    if (lane < n) out[(size_t)b * n + lane] = ex / sm;
}

// ---------------------------------------------------------------------------
// out16[b,m] = sum_e wl[b,e]*mloc[b,e,m] + sum_j wg[b,j]*cand[b,j,m]
// inputs f32, output f16 (feeds downstream GEMM A / LoRA reduce)
// ---------------------------------------------------------------------------
__global__ void combine_gates(const float* __restrict__ wl, int ne,
                              const float* __restrict__ mloc,
                              const float* __restrict__ wg,
                              const float* __restrict__ cA,
                              const float* __restrict__ cB,
                              const float* __restrict__ cC,
                              int nc, int Mdim, _Float16* __restrict__ out,
                              long total)
{
    long i = (long)blockIdx.x * 256 + threadIdx.x;
    if (i >= total) return;
    long b = i / Mdim; int m = (int)(i - b * Mdim);
    float acc = 0.f;
    for (int e = 0; e < ne; ++e)
        acc = fmaf(wl[b * ne + e], mloc[((size_t)b * ne + e) * Mdim + m], acc);
    for (int j = 0; j < nc; ++j) {
        int blk = j / ne, jj = j - blk * ne;
        const float* src = (blk == 0) ? cA : ((blk == 1) ? cB : cC);
        acc = fmaf(wg[b * nc + j], src[((size_t)b * ne + jj) * Mdim + m], acc);
    }
    out[i] = (_Float16)acc;
}

// ---------------------------------------------------------------------------
// Host orchestration
// ---------------------------------------------------------------------------
extern "C" void kernel_launch(void* const* d_in, const int* in_sizes, int n_in,
                              void* d_out, int out_size, void* d_ws, size_t ws_size,
                              hipStream_t stream)
{
    (void)in_sizes; (void)n_in; (void)out_size; (void)ws_size;
    const float* x    = (const float*)d_in[0];  // [6,1024,512]
    const float* lin  = (const float*)d_in[1];  // [1024,128]
    const float* gin  = (const float*)d_in[2];  // [1024,128]
    const float* ltin = (const float*)d_in[3];  // [1024,128]
    auto F = [&](int i) { return (const float*)d_in[i]; };
    // params flattened in insertion order:
    //  groups[gi] @ 4+gi*12 : Af,Bf,W1,b1,W2,b2,g1,bt1,Wl,bl,Wg,bg
    //  local[i]   @ 40+i*6  : Af,Bf,W,b,g,bt
    //  tasks[t]   @ 52+t*10 : W,b,g,bt,Wl,bl,Wg,bg,Af,Bf

    float* ws = (float*)d_ws;
    size_t off = 0;
    auto allocF = [&](size_t n) { float* p = ws + off; off += n; return p; };
    auto allocH = [&](size_t nh) {
        _Float16* p = (_Float16*)(ws + off); off += (nh + 1) / 2; return p;
    };
    float* meta[3];
    for (int i = 0; i < 3; ++i) meta[i] = allocF((size_t)BSZ * 8 * 512);
    _Float16* Wt16   = allocH((size_t)8 * 1536 * 768);   // max weight (g0 W1)
    _Float16* vbuf16 = allocH((size_t)BSZ * 1536);
    _Float16* hbuf16 = allocH((size_t)BSZ * 8 * 768);
    _Float16* gout16 = allocH((size_t)3 * BSZ * 512);
    _Float16* mix16  = allocH((size_t)BSZ * 256);
    float* tvec  = allocF(2 * BSZ);
    float* bnsum = allocF(4096);
    float* bnsq  = allocF(4096);
    float* wlb   = allocF((size_t)BSZ * 8);
    float* wgb   = allocF((size_t)BSZ * 24);
    float* Lb    = allocF((size_t)2 * BSZ * 4 * 256);
    float* Te    = allocF((size_t)BSZ * 4 * 256);
    float* Se    = allocF((size_t)BSZ * 4 * 256);

    const int gDg[3]   = {1536, 1024, 512};
    const int gHid[3]  = {768, 512, 512};
    const int gT[3][3] = {{0, 1, 2}, {3, 4, 0}, {5, 0, 0}};

    // ---- meta layer: per-group LoRA gate -> 2-layer relu bank -> BN+SiLU ----
    for (int gi = 0; gi < 3; ++gi) {
        int base = 4 + gi * 12;
        const float *Af = F(base+0), *Bf = F(base+1), *W1 = F(base+2),
                    *b1 = F(base+3), *W2 = F(base+4), *b2 = F(base+5),
                    *g1 = F(base+6), *bt1 = F(base+7);
        int Dg = gDg[gi], hid = gHid[gi];
        lora_rank2_reduce<<<BSZ, 32, 0, stream>>>(
            nullptr, 0, x, gT[gi][0], gT[gi][1], gT[gi][2], 1, Af, Dg, tvec);
        long tv = (long)BSZ * Dg;
        lora_rank2_apply<_Float16><<<(int)((tv + 255) / 256), 256, 0, stream>>>(
            tvec, Bf, vbuf16, Dg, tv);
        // layer 1
        long wtot = 8L * Dg * hid;
        cvt_w_transpose<<<(int)((wtot + 255) / 256), 256, 0, stream>>>(
            W1, Wt16, Dg, hid, wtot);
        gemm_bias_relu<<<dim3(hid / TBN, BSZ / TBM, 8), 256, 0, stream>>>(
            vbuf16, Wt16, b1, nullptr, hbuf16, Dg, hid, Dg, 0L, 1);
        // layer 2
        wtot = 8L * hid * 512;
        cvt_w_transpose<<<(int)((wtot + 255) / 256), 256, 0, stream>>>(
            W2, Wt16, hid, 512, wtot);
        gemm_bias_relu<<<dim3(512 / TBN, BSZ / TBM, 8), 256, 0, stream>>>(
            hbuf16, Wt16, b2, meta[gi], nullptr, hid, 512, 8 * hid, (long)hid, 0);
        bn_stats<<<(8 * 512 + 255) / 256, 256, 0, stream>>>(
            meta[gi], bnsum, bnsq, 8 * 512);
        long tm = (long)BSZ * 8 * 512;
        bn_apply_silu<<<(int)((tm + 255) / 256), 256, 0, stream>>>(
            meta[gi], bnsum, bnsq, g1, bt1, 8 * 512, tm);
    }

    // ---- meta gating -> group_out (f16) ----
    for (int gi = 0; gi < 3; ++gi) {
        int base = 4 + gi * 12;
        const float *Wl = F(base+8), *bl = F(base+9),
                    *Wg = F(base+10), *bg = F(base+11);
        gate_softmax<<<BSZ, 32, 0, stream>>>(lin, Wl, bl, 8, wlb);
        int nc = (gi == 2) ? 24 : 16;
        gate_softmax<<<BSZ, 32, 0, stream>>>(gin, Wg, bg, nc, wgb);
        const float *cA, *cB, *cC;
        if (gi == 0)      { cA = meta[0]; cB = meta[2]; cC = meta[2]; }
        else if (gi == 1) { cA = meta[1]; cB = meta[2]; cC = meta[2]; }
        else              { cA = meta[0]; cB = meta[1]; cC = meta[2]; }
        long tg = (long)BSZ * 512;
        combine_gates<<<(int)((tg + 255) / 256), 256, 0, stream>>>(
            wlb, 8, meta[gi], wgb, cA, cB, cC, nc, 512,
            gout16 + (size_t)gi * BSZ * 512, tg);
    }

    // ---- local expert banks (with group LoRA gate) ----
    for (int i = 0; i < 2; ++i) {
        int base = 40 + i * 6;
        const float *Af = F(base+0), *Bf = F(base+1), *W = F(base+2),
                    *bb = F(base+3), *g = F(base+4), *bt = F(base+5);
        const _Float16* gi16 = gout16 + (size_t)i * BSZ * 512;
        lora_rank2_reduce<<<BSZ, 32, 0, stream>>>(
            gi16, 512, nullptr, 0, 0, 0, 0, Af, 512, tvec);
        long tv = (long)BSZ * 512;
        lora_rank2_apply<_Float16><<<(int)((tv + 255) / 256), 256, 0, stream>>>(
            tvec, Bf, vbuf16, 512, tv);
        long wtot = 4L * 512 * 256;
        cvt_w_transpose<<<(int)((wtot + 255) / 256), 256, 0, stream>>>(
            W, Wt16, 512, 256, wtot);
        float* Li = Lb + (size_t)i * BSZ * 4 * 256;
        gemm_bias_relu<<<dim3(256 / TBN, BSZ / TBM, 4), 256, 0, stream>>>(
            vbuf16, Wt16, bb, Li, nullptr, 512, 256, 512, 0L, 0);
        bn_stats<<<(4 * 256 + 255) / 256, 256, 0, stream>>>(Li, bnsum, bnsq, 1024);
        long tm = (long)BSZ * 1024;
        bn_apply_silu<<<(int)((tm + 255) / 256), 256, 0, stream>>>(
            Li, bnsum, bnsq, g, bt, 1024, tm);
    }

    // ---- task layer ----
    const int giOf[6] = {0, 0, 0, 1, 1, 2};
    float* outp = (float*)d_out;  // [6, 1024, 256] f32
    for (int t = 0; t < 6; ++t) {
        int base = 52 + t * 10;
        const float *W = F(base+0), *bb = F(base+1), *g = F(base+2),
                    *bt = F(base+3), *Wl = F(base+4), *bl = F(base+5),
                    *Wg = F(base+6), *bg = F(base+7),
                    *Af = F(base+8), *Bf = F(base+9);
        int gi = giOf[t];
        int Lidx = (gi < 1) ? 0 : 1;   // min(gi, 1)
        const _Float16* gsrc = gout16 + (size_t)gi * BSZ * 512;
        const _Float16* ssrc = gout16 + (size_t)2 * BSZ * 512;

        long wtot = 4L * 512 * 256;
        cvt_w_transpose<<<(int)((wtot + 255) / 256), 256, 0, stream>>>(
            W, Wt16, 512, 256, wtot);

        // Te = bank(group_out[gi]) ; Se = bank(group_out[share]) (same weights)
        gemm_bias_relu<<<dim3(2, BSZ / TBM, 4), 256, 0, stream>>>(
            gsrc, Wt16, bb, Te, nullptr, 512, 256, 512, 0L, 0);
        bn_stats<<<4, 256, 0, stream>>>(Te, bnsum, bnsq, 1024);
        long tm = (long)BSZ * 1024;
        bn_apply_silu<<<(int)((tm + 255) / 256), 256, 0, stream>>>(
            Te, bnsum, bnsq, g, bt, 1024, tm);

        gemm_bias_relu<<<dim3(2, BSZ / TBM, 4), 256, 0, stream>>>(
            ssrc, Wt16, bb, Se, nullptr, 512, 256, 512, 0L, 0);
        bn_stats<<<4, 256, 0, stream>>>(Se, bnsum, bnsq, 1024);
        bn_apply_silu<<<(int)((tm + 255) / 256), 256, 0, stream>>>(
            Se, bnsum, bnsq, g, bt, 1024, tm);

        gate_softmax<<<BSZ, 32, 0, stream>>>(ltin, Wl, bl, 4, wlb);
        gate_softmax<<<BSZ, 32, 0, stream>>>(gin, Wg, bg, 12, wgb);

        const float* Li = Lb + (size_t)Lidx * BSZ * 4 * 256;
        long to = (long)BSZ * 256;
        combine_gates<<<(int)((to + 255) / 256), 256, 0, stream>>>(
            wlb, 4, Te, wgb, Te, Li, Se, 12, 256, mix16, to);

        lora_rank2_reduce<<<BSZ, 32, 0, stream>>>(
            mix16, 256, nullptr, 0, 0, 0, 0, Af, 256, tvec);
        lora_rank2_apply<float><<<(int)((to + 255) / 256), 256, 0, stream>>>(
            tvec, Bf, outp + (size_t)t * BSZ * 256, 256, to);
    }
}
